// PointUNet_89481348644933
// MI455X (gfx1250) — matrix operand order repaired and checked by reading, not compiled, not run on hardware
//
#include <hip/hip_runtime.h>
#include <math.h>

// ---------------------------------------------------------------------------
// PointUNet forward for MI455X (gfx1250, wave32, WMMA).
// All matmuls run on v_wmma_f32_16x16x32_f16 (f16 in, f32 accum).
// Weights are converted to K-padded f16 once per launch; activations are f16
// end-to-end, so every WMMA fragment is exactly two b128 loads per lane.
// FP GEMMs: 2 N-tiles per wave (shared A fragment) + software-pipelined
// double-buffered loads so v_wmma overlaps the next chunk's global_load_b128.
// SA MLPs: fused gather->3 layers->maxpool per ball-query group in LDS,
// 2 N-tiles per wave sharing one ds_load_b128 A fragment.
// Requires ~112 MB of d_ws.
// ---------------------------------------------------------------------------

typedef __attribute__((ext_vector_type(16))) _Float16 v16h;
typedef __attribute__((ext_vector_type(8)))  _Float16 v8h;
typedef __attribute__((ext_vector_type(8)))  float    v8f;

union H16 { v16h v; v8h h[2]; };

#define BATCH 8
#define NPTS  8192

// ---------------------------------------------------------------------------
// f32 -> f16 weight conversion with K padding to multiple of 32 (zeros).
// ---------------------------------------------------------------------------
__global__ void __launch_bounds__(256) k_cvt_w(
    const float* __restrict__ W, _Float16* __restrict__ out,
    int O, int K, int KP)
{
  const int i = blockIdx.x * blockDim.x + threadIdx.x;
  if (i >= O * KP) return;
  const int o = i / KP, k = i % KP;
  out[i] = (k < K) ? (_Float16)W[(size_t)o * K + k] : (_Float16)0.0f;
}

// ---------------------------------------------------------------------------
// Positional embedding + 2x SiLU linear (tiny: B=8) -> te (8,512) f32
// ---------------------------------------------------------------------------
__global__ void __launch_bounds__(256) k_pos_emb(
    const float* __restrict__ tt,
    const float* __restrict__ W0, const float* __restrict__ B0,
    const float* __restrict__ W1, const float* __restrict__ B1,
    float* __restrict__ te)
{
  __shared__ float e0[BATCH * 256];
  __shared__ float e1[BATCH * 512];
  const int tid = threadIdx.x;
  for (int i = tid; i < BATCH * 256; i += 256) {
    int b = i / 256, c = i % 256;
    float freq = powf(10000.0f, -(float)(c & 127) / 128.0f);
    float ang  = tt[b] * freq;
    e0[i] = (c < 128) ? cosf(ang) : sinf(ang);
  }
  __syncthreads();
  for (int i = tid; i < BATCH * 512; i += 256) {
    int b = i / 512, o = i % 512;
    float acc = B0[o];
    for (int k = 0; k < 256; ++k) acc += e0[b * 256 + k] * W0[o * 256 + k];
    e1[i] = acc / (1.0f + expf(-acc));
  }
  __syncthreads();
  for (int i = tid; i < BATCH * 512; i += 256) {
    int b = i / 512, o = i % 512;
    float acc = B1[o];
    for (int k = 0; k < 512; ++k) acc += e1[b * 512 + k] * W1[o * 512 + k];
    te[i] = acc / (1.0f + expf(-acc));
  }
}

// ---------------------------------------------------------------------------
// Farthest point sampling: one block per batch, sequential over npoint.
// ---------------------------------------------------------------------------
__global__ void __launch_bounds__(256) k_fps(
    const float* __restrict__ xyz, int N, int npoint,
    int* __restrict__ fidx, float* __restrict__ new_xyz)
{
  const int b   = blockIdx.x;
  const int tid = threadIdx.x;
  const float* pts = xyz + (size_t)b * N * 3;
  __shared__ float dist[NPTS];
  __shared__ float rv[256];
  __shared__ int   ri[256];
  __shared__ int   sfar;
  for (int i = tid; i < N; i += 256) dist[i] = 1.0e10f;
  if (tid == 0) sfar = 0;
  __syncthreads();
  for (int j = 0; j < npoint; ++j) {
    const int far = sfar;
    if (tid == 0) {
      fidx[(size_t)b * npoint + j] = far;
      new_xyz[((size_t)b * npoint + j) * 3 + 0] = pts[far * 3 + 0];
      new_xyz[((size_t)b * npoint + j) * 3 + 1] = pts[far * 3 + 1];
      new_xyz[((size_t)b * npoint + j) * 3 + 2] = pts[far * 3 + 2];
    }
    const float fx = pts[far * 3 + 0], fy = pts[far * 3 + 1], fz = pts[far * 3 + 2];
    float best = -1.0f; int bi = 0;
    for (int i = tid; i < N; i += 256) {
      float dx = pts[i * 3 + 0] - fx;
      float dy = pts[i * 3 + 1] - fy;
      float dz = pts[i * 3 + 2] - fz;
      float d  = dx * dx + dy * dy + dz * dz;
      float dm = fminf(dist[i], d);
      dist[i] = dm;
      if (dm > best) { best = dm; bi = i; }
    }
    rv[tid] = best; ri[tid] = bi;
    __syncthreads();
    for (int off = 128; off > 0; off >>= 1) {
      if (tid < off) {
        float ov = rv[tid + off]; int oi = ri[tid + off];
        if (ov > rv[tid] || (ov == rv[tid] && oi < ri[tid])) { rv[tid] = ov; ri[tid] = oi; }
      }
      __syncthreads();
    }
    if (tid == 0) sfar = ri[0];
    __syncthreads();
  }
}

// ---------------------------------------------------------------------------
// Ball query: one wave32 per query; first `nsample` in-radius indices in
// index order (ballot_w32 + prefix popcount), padded with first found.
// ---------------------------------------------------------------------------
__global__ void __launch_bounds__(32) k_ballquery(
    const float* __restrict__ xyz, int N,
    const float* __restrict__ new_xyz, int S,
    float r2, int nsample, int* __restrict__ out_idx)
{
  const int q = blockIdx.x;          // b*S + s
  const int b = q / S;
  const int lane = threadIdx.x;
  const float* pts = xyz + (size_t)b * N * 3;
  const float qx = new_xyz[(size_t)q * 3 + 0];
  const float qy = new_xyz[(size_t)q * 3 + 1];
  const float qz = new_xyz[(size_t)q * 3 + 2];
  int* out = out_idx + (size_t)q * nsample;
  int cnt = 0, first = -1;
  for (int base = 0; base < N && cnt < nsample; base += 32) {
    const int i = base + lane;
    const float dx = pts[i * 3 + 0] - qx;
    const float dy = pts[i * 3 + 1] - qy;
    const float dz = pts[i * 3 + 2] - qz;
    const bool inr = (dx * dx + dy * dy + dz * dz) < r2;
    const unsigned m = __builtin_amdgcn_ballot_w32(inr);
    if (inr) {
      const int pos = cnt + __popc(m & ((1u << lane) - 1u));
      if (pos < nsample) out[pos] = i;
    }
    if (first < 0 && m != 0u) first = base + __ffs((int)m) - 1;
    cnt += __popc(m);
  }
  if (first < 0) first = 0;
  const int c0 = cnt < nsample ? cnt : nsample;
  for (int p = c0 + lane; p < nsample; p += 32) out[p] = first;
}

// ---------------------------------------------------------------------------
// One shared-MLP layer on an LDS tile: out = relu(in * W^T + b).
// Two adjacent 16-col output tiles per wave share one A fragment
// (two ds_load_b128 per 32-K step feed two v_wmma).
// ---------------------------------------------------------------------------
template <int MT, int K, int LDIN, int COUT, int LDOUT>
__device__ __forceinline__ void sa_layer(const _Float16* in, _Float16* out,
                                         const _Float16* __restrict__ W,
                                         const float* __restrict__ Bv)
{
  constexpr int KP = (K + 31) & ~31;
  static_assert(LDIN >= KP, "input tile must be padded");
  static_assert(MT % 16 == 0 && COUT % 32 == 0, "tile dims");
  static_assert(LDIN % 8 == 0 && LDOUT % 8 == 0, "b128 alignment");
  const int lane = threadIdx.x & 31;
  const int wave = threadIdx.x >> 5;
  const int ln   = lane & 15;
  const int half = lane >> 4;
  constexpr int TM = MT / 16, TN2 = COUT / 32, NT = TM * TN2;
  for (int t = wave; t < NT; t += 4) {
    const int tm = t / TN2;
    const int tn = (t % TN2) * 2;
    const int m  = tm * 16 + ln;
    const int o0 = tn * 16 + ln;
    const int o1 = o0 + 16;
    const _Float16* w0 = W + (size_t)o0 * KP;
    const _Float16* w1 = W + (size_t)o1 * KP;
    v8f acc0 = {}, acc1 = {};
    for (int kk = 0; kk < KP; kk += 32) {
      H16 a, b0, b1;
      a.h[0]  = *(const v8h*)(in + m * LDIN + kk + half * 16);
      a.h[1]  = *(const v8h*)(in + m * LDIN + kk + half * 16 + 8);
      b0.h[0] = *(const v8h*)(w0 + kk + half * 16);
      b0.h[1] = *(const v8h*)(w0 + kk + half * 16 + 8);
      b1.h[0] = *(const v8h*)(w1 + kk + half * 16);
      b1.h[1] = *(const v8h*)(w1 + kk + half * 16 + 8);
      acc0 = __builtin_amdgcn_wmma_f32_16x16x32_f16(false, a.v, false, b0.v,
                                                    (short)0, acc0, false, false);
      acc1 = __builtin_amdgcn_wmma_f32_16x16x32_f16(false, a.v, false, b1.v,
                                                    (short)0, acc1, false, false);
    }
    const float bias0 = Bv[o0];
    const float bias1 = Bv[o1];
#pragma unroll
    for (int r = 0; r < 8; ++r) {
      const int mm = tm * 16 + half * 8 + r;
      float v0 = acc0[r] + bias0; v0 = v0 > 0.0f ? v0 : 0.0f;
      float v1 = acc1[r] + bias1; v1 = v1 > 0.0f ? v1 : 0.0f;
      out[mm * LDOUT + o0] = (_Float16)v0;
      out[mm * LDOUT + o1] = (_Float16)v1;
    }
  }
}

// ---------------------------------------------------------------------------
// Fused SA stage: gather group into LDS -> 3 WMMA MLP layers -> maxpool.
// One block (4 waves) per (b,s) group.  feats/outf are f16.
// ---------------------------------------------------------------------------
template <int NS, int C0, int C1, int C2, int C3>
__global__ void __launch_bounds__(128) k_sa_mlp(
    const float* __restrict__ xyz, int N,
    const float* __restrict__ new_xyz, int S,
    const _Float16* __restrict__ feats,          // N x (C0-3), may be null
    const int* __restrict__ gidx, float inv_radius,
    const _Float16* __restrict__ W1, const float* __restrict__ B1,
    const _Float16* __restrict__ W2, const float* __restrict__ B2,
    const _Float16* __restrict__ W3, const float* __restrict__ B3,
    _Float16* __restrict__ outf)                 // (B*S) x C3
{
  constexpr int MT  = NS < 16 ? 16 : NS;
  constexpr int C0P = (C0 + 31) & ~31;
  constexpr int LDA = C0P > C2 ? C0P : C2;       // input tile, then layer2 out
  constexpr int LDB = C1 > C3 ? C1 : C3;         // layer1 out, then layer3 out
  __shared__ __align__(16) _Float16 bufA[MT * LDA];
  __shared__ __align__(16) _Float16 bufB[MT * LDB];
  const int q   = blockIdx.x;                    // b*S + s
  const int b   = q / S;
  const int tid = threadIdx.x;
  for (int i = tid; i < MT * LDA; i += 128) bufA[i] = (_Float16)0.0f;
  __syncthreads();
  const int* gi = gidx + (size_t)q * NS;
  for (int e = tid; e < NS * C0; e += 128) {
    const int j = e / C0, c = e % C0;
    const int pi = gi[j];
    _Float16 v;
    if (c < 3) v = (_Float16)((xyz[((size_t)b * N + pi) * 3 + c] -
                               new_xyz[(size_t)q * 3 + c]) * inv_radius);
    else       v = feats[((size_t)b * N + pi) * (size_t)(C0 - 3) + (c - 3)];
    bufA[j * LDA + c] = v;
  }
  __syncthreads();
  sa_layer<MT, C0, LDA, C1, C1>(bufA, bufB, W1, B1);
  __syncthreads();
  sa_layer<MT, C1, C1, C2, C2>(bufB, bufA, W2, B2);
  __syncthreads();
  sa_layer<MT, C2, C2, C3, C3>(bufA, bufB, W3, B3);
  __syncthreads();
  for (int c = tid; c < C3; c += 128) {
    float mx = -3.0e38f;
    for (int j = 0; j < NS; ++j) mx = fmaxf(mx, (float)bufB[j * C3 + c]);
    outf[(size_t)q * C3 + c] = (_Float16)mx;
  }
}

// ---------------------------------------------------------------------------
// Generic WMMA linear: Y[M,O] = relu(X[M,K] @ W[O,K]^T + b), f16 in/out.
// Two adjacent N-tiles per wave (shared A fragment) + software-pipelined
// double-buffered fragments: while v_wmma consumes chunk k, the b128 loads
// for chunk k+1 are already in flight (partial s_wait_loadcnt).
// M%16==0, K%32==0, O%128==0.  grid = (O/128, M/16), block = 128.
// ---------------------------------------------------------------------------
__global__ void __launch_bounds__(128) k_linear_relu(
    const _Float16* __restrict__ X, const _Float16* __restrict__ W,
    const float* __restrict__ Bv, _Float16* __restrict__ Y,
    int M, int K, int O)
{
  const int lane = threadIdx.x & 31;
  const int wave = threadIdx.x >> 5;
  const int ln   = lane & 15;
  const int half = lane >> 4;
  const int tm   = blockIdx.y;
  const int tn   = (blockIdx.x * 4 + wave) * 2;
  const int m    = tm * 16 + ln;
  const int o0   = tn * 16 + ln;
  const int o1   = o0 + 16;
  const _Float16* xr = X + (size_t)m * K;
  const _Float16* w0 = W + (size_t)o0 * K;
  const _Float16* w1 = W + (size_t)o1 * K;
  v8f acc0 = {}, acc1 = {};
  H16 a, b0, b1;
  a.h[0]  = *(const v8h*)(xr + half * 16);
  a.h[1]  = *(const v8h*)(xr + half * 16 + 8);
  b0.h[0] = *(const v8h*)(w0 + half * 16);
  b0.h[1] = *(const v8h*)(w0 + half * 16 + 8);
  b1.h[0] = *(const v8h*)(w1 + half * 16);
  b1.h[1] = *(const v8h*)(w1 + half * 16 + 8);
  for (int kk = 32; kk < K; kk += 32) {
    __builtin_prefetch(xr + kk + 32, 0, 3);
    __builtin_prefetch(w0 + kk + 32, 0, 3);
    __builtin_prefetch(w1 + kk + 32, 0, 3);
    H16 an, b0n, b1n;
    an.h[0]  = *(const v8h*)(xr + kk + half * 16);
    an.h[1]  = *(const v8h*)(xr + kk + half * 16 + 8);
    b0n.h[0] = *(const v8h*)(w0 + kk + half * 16);
    b0n.h[1] = *(const v8h*)(w0 + kk + half * 16 + 8);
    b1n.h[0] = *(const v8h*)(w1 + kk + half * 16);
    b1n.h[1] = *(const v8h*)(w1 + kk + half * 16 + 8);
    acc0 = __builtin_amdgcn_wmma_f32_16x16x32_f16(false, a.v, false, b0.v,
                                                  (short)0, acc0, false, false);
    acc1 = __builtin_amdgcn_wmma_f32_16x16x32_f16(false, a.v, false, b1.v,
                                                  (short)0, acc1, false, false);
    a = an; b0 = b0n; b1 = b1n;
  }
  acc0 = __builtin_amdgcn_wmma_f32_16x16x32_f16(false, a.v, false, b0.v,
                                                (short)0, acc0, false, false);
  acc1 = __builtin_amdgcn_wmma_f32_16x16x32_f16(false, a.v, false, b1.v,
                                                (short)0, acc1, false, false);
  const float bias0 = Bv[o0];
  const float bias1 = Bv[o1];
#pragma unroll
  for (int r = 0; r < 8; ++r) {
    const int mm = tm * 16 + half * 8 + r;
    float v0 = acc0[r] + bias0; v0 = v0 > 0.0f ? v0 : 0.0f;
    float v1 = acc1[r] + bias1; v1 = v1 > 0.0f ? v1 : 0.0f;
    Y[(size_t)mm * O + o0] = (_Float16)v0;
    Y[(size_t)mm * O + o1] = (_Float16)v1;
  }
}

// ---------------------------------------------------------------------------
// three_nn + inverse-distance interpolate + concat(uf [+te]) -> X (f16).
// One thread per up-sampled point.
// ---------------------------------------------------------------------------
__global__ void __launch_bounds__(256) k_three_interp(
    const float* __restrict__ ux, int nu,
    const float* __restrict__ kx, int nk,
    const _Float16* __restrict__ kf, int Ck,
    const _Float16* __restrict__ uf, int Cu,
    const float* __restrict__ te, int addte,
    _Float16* __restrict__ X)
{
  const int g = blockIdx.x * blockDim.x + threadIdx.x;
  const int total = BATCH * nu;
  if (g >= total) return;
  const int b = g / nu;
  const float px = ux[(size_t)g * 3 + 0];
  const float py = ux[(size_t)g * 3 + 1];
  const float pz = ux[(size_t)g * 3 + 2];
  const float* kp = kx + (size_t)b * nk * 3;
  float d0 = 1e30f, d1 = 1e30f, d2 = 1e30f;
  int   i0 = 0, i1 = 0, i2 = 0;
  for (int i = 0; i < nk; ++i) {
    float dx = px - kp[i * 3 + 0];
    float dy = py - kp[i * 3 + 1];
    float dz = pz - kp[i * 3 + 2];
    float d  = dx * dx + dy * dy + dz * dz;
    if (d < d0)      { d2 = d1; i2 = i1; d1 = d0; i1 = i0; d0 = d; i0 = i; }
    else if (d < d1) { d2 = d1; i2 = i1; d1 = d;  i1 = i; }
    else if (d < d2) { d2 = d;  i2 = i; }
  }
  float w0 = 1.0f / (fmaxf(d0, 0.0f) + 1e-8f);
  float w1 = 1.0f / (fmaxf(d1, 0.0f) + 1e-8f);
  float w2 = 1.0f / (fmaxf(d2, 0.0f) + 1e-8f);
  const float ws = w0 + w1 + w2;
  w0 /= ws; w1 /= ws; w2 /= ws;
  const int Cin = Ck + Cu;
  _Float16* xo = X + (size_t)g * Cin;
  const _Float16* f0 = kf + ((size_t)b * nk + i0) * Ck;
  const _Float16* f1 = kf + ((size_t)b * nk + i1) * Ck;
  const _Float16* f2 = kf + ((size_t)b * nk + i2) * Ck;
  for (int c = 0; c < Ck; ++c)
    xo[c] = (_Float16)(w0 * (float)f0[c] + w1 * (float)f1[c] + w2 * (float)f2[c]);
  if (Cu > 0) {
    const _Float16* u0 = uf + (size_t)g * Cu;
    if (addte) { for (int c = 0; c < Cu; ++c) xo[Ck + c] = (_Float16)((float)u0[c] + te[b * 512 + c]); }
    else       { for (int c = 0; c < Cu; ++c) xo[Ck + c] = u0[c]; }
  }
}

// ---------------------------------------------------------------------------
// LayerNorm(128) + out projection (3x128).  f16 in, f32 out.
// ---------------------------------------------------------------------------
__global__ void __launch_bounds__(256) k_ln_out(
    const _Float16* __restrict__ f, const float* __restrict__ g,
    const float* __restrict__ be, const float* __restrict__ ow,
    float* __restrict__ out, int M)
{
  const int i = blockIdx.x * blockDim.x + threadIdx.x;
  if (i >= M) return;
  const _Float16* x = f + (size_t)i * 128;
  float mu = 0.0f;
  for (int c = 0; c < 128; ++c) mu += (float)x[c];
  mu *= (1.0f / 128.0f);
  float var = 0.0f;
  for (int c = 0; c < 128; ++c) { float d = (float)x[c] - mu; var += d * d; }
  var *= (1.0f / 128.0f);
  const float inv = rsqrtf(var + 1e-5f);
  float o0 = 0.0f, o1 = 0.0f, o2 = 0.0f;
  for (int c = 0; c < 128; ++c) {
    const float h = ((float)x[c] - mu) * inv * g[c] + be[c];
    o0 += h * ow[0 * 128 + c];
    o1 += h * ow[1 * 128 + c];
    o2 += h * ow[2 * 128 + c];
  }
  out[(size_t)i * 3 + 0] = o0;
  out[(size_t)i * 3 + 1] = o1;
  out[(size_t)i * 3 + 2] = o2;
}

// ---------------------------------------------------------------------------
// Host orchestration.  Input order (insertion order of setup_inputs):
//   0: noise_points  1: t  2: cond_points
//   3..26 : noisy SA (4 levels x 3 layers x {W,b})
//   27..50: cond  SA
//   51..82: fp in order fp1_cross,fp1,fp2_cross,fp2,fp3_cross,fp3,fp4_cross,fp4
//   83,84: map0 {W,b}  85,86: map1 {W,b}  87,88: ln {g,b}  89: out_w
// ---------------------------------------------------------------------------
extern "C" void kernel_launch(void* const* d_in, const int* in_sizes, int n_in,
                              void* d_out, int out_size, void* d_ws, size_t ws_size,
                              hipStream_t stream)
{
  (void)in_sizes; (void)n_in; (void)out_size; (void)ws_size;
  const float* noise_pts = (const float*)d_in[0];
  const float* tptr      = (const float*)d_in[1];
  const float* cond_pts  = (const float*)d_in[2];

  int ip = 3;
  const float *nW[4][3], *nB[4][3], *cW[4][3], *cB[4][3];
  for (int l = 0; l < 4; ++l) for (int j = 0; j < 3; ++j) {
    nW[l][j] = (const float*)d_in[ip++]; nB[l][j] = (const float*)d_in[ip++];
  }
  for (int l = 0; l < 4; ++l) for (int j = 0; j < 3; ++j) {
    cW[l][j] = (const float*)d_in[ip++]; cB[l][j] = (const float*)d_in[ip++];
  }
  const float *fW[8][2], *fB[8][2];
  for (int f = 0; f < 8; ++f) for (int j = 0; j < 2; ++j) {
    fW[f][j] = (const float*)d_in[ip++]; fB[f][j] = (const float*)d_in[ip++];
  }
  const float* m0W = (const float*)d_in[ip++];
  const float* m0B = (const float*)d_in[ip++];
  const float* m1W = (const float*)d_in[ip++];
  const float* m1B = (const float*)d_in[ip++];
  const float* lnG = (const float*)d_in[ip++];
  const float* lnO = (const float*)d_in[ip++];
  const float* oW  = (const float*)d_in[ip++];

  // ---- workspace carve ----
  size_t off = 0;
  auto AF = [&](size_t nfloats) -> float* {
    float* p = (float*)((char*)d_ws + off);
    off += (nfloats * sizeof(float) + 255) & ~(size_t)255;
    return p;
  };
  auto AH = [&](size_t nhalf) -> _Float16* {
    _Float16* p = (_Float16*)((char*)d_ws + off);
    off += (nhalf * sizeof(_Float16) + 255) & ~(size_t)255;
    return p;
  };
  float* te = AF(BATCH * 512);
  float *nx1 = AF(BATCH*2048*3), *nx2 = AF(BATCH*1024*3);
  float *nx3 = AF(BATCH*512*3),  *nx4 = AF(BATCH*256*3);
  float *cx1 = AF(BATCH*2048*3), *cx2 = AF(BATCH*1024*3);
  float *cx3 = AF(BATCH*512*3),  *cx4 = AF(BATCH*256*3);
  _Float16 *nf1 = AH((size_t)BATCH*2048*128), *nf2 = AH((size_t)BATCH*1024*256);
  _Float16 *nf3 = AH((size_t)BATCH*512*512),  *nf4 = AH((size_t)BATCH*256*512);
  _Float16 *cf1 = AH((size_t)BATCH*2048*128), *cf2 = AH((size_t)BATCH*1024*256);
  _Float16 *cf3 = AH((size_t)BATCH*512*512),  *cf4 = AH((size_t)BATCH*256*512);
  int*   fidx = (int*)AF(BATCH * 2048);
  int*   gidx = (int*)AF((size_t)BATCH * 2048 * 64);
  _Float16* xbuf = AH((size_t)65536 * 128);   // max M*K over FP stages
  _Float16* hbuf = AH((size_t)65536 * 128);
  _Float16* fAb  = AH((size_t)65536 * 128);
  _Float16* fBb  = AH((size_t)65536 * 128);

  // ---- weight conversion (f32 -> f16, K padded to %32) ----
  auto cvt = [&](const float* W, int O, int K) -> _Float16* {
    const int KP = (K + 31) & ~31;
    _Float16* p = AH((size_t)O * KP);
    const int total = O * KP;
    k_cvt_w<<<(total + 255) / 256, 256, 0, stream>>>(W, p, O, K, KP);
    return p;
  };
  static const int saK[4][3] = {{3,64,64},{131,128,128},{259,256,256},{515,512,512}};
  static const int saO[4][3] = {{64,64,128},{128,128,256},{256,256,512},{512,512,512}};
  _Float16 *nWh[4][3], *cWh[4][3];
  for (int l = 0; l < 4; ++l) for (int j = 0; j < 3; ++j) {
    nWh[l][j] = cvt(nW[l][j], saO[l][j], saK[l][j]);
    cWh[l][j] = cvt(cW[l][j], saO[l][j], saK[l][j]);
  }
  static const int fpK[8][2] = {{1024,512},{1024,512},{1024,512},{512,512},
                                {512,256},{256,256},{256,128},{128,128}};
  static const int fpO[8][2] = {{512,512},{512,512},{512,256},{512,256},
                                {256,128},{128,128},{128,128},{128,128}};
  _Float16* fWh[8][2];
  for (int f = 0; f < 8; ++f) for (int j = 0; j < 2; ++j)
    fWh[f][j] = cvt(fW[f][j], fpO[f][j], fpK[f][j]);

  // ---- time embedding ----
  k_pos_emb<<<1, 256, 0, stream>>>(tptr, m0W, m0B, m1W, m1B, te);

  // ---- encoders ----
  auto encode = [&](const float* pts,
                    _Float16* (*Wh)[3], const float* (*Bs)[3],
                    float* x1, _Float16* f1, float* x2, _Float16* f2,
                    float* x3, _Float16* f3, float* x4, _Float16* f4) {
    k_fps<<<BATCH, 256, 0, stream>>>(pts, 8192, 2048, fidx, x1);
    k_ballquery<<<BATCH * 2048, 32, 0, stream>>>(pts, 8192, x1, 2048, 0.04f, 64, gidx);
    k_sa_mlp<64, 3, 64, 64, 128><<<BATCH * 2048, 128, 0, stream>>>(
        pts, 8192, x1, 2048, nullptr, gidx, 1.0f / 0.2f,
        Wh[0][0], Bs[0][0], Wh[0][1], Bs[0][1], Wh[0][2], Bs[0][2], f1);
    k_fps<<<BATCH, 256, 0, stream>>>(x1, 2048, 1024, fidx, x2);
    k_ballquery<<<BATCH * 1024, 32, 0, stream>>>(x1, 2048, x2, 1024, 0.16f, 32, gidx);
    k_sa_mlp<32, 131, 128, 128, 256><<<BATCH * 1024, 128, 0, stream>>>(
        x1, 2048, x2, 1024, f1, gidx, 1.0f / 0.4f,
        Wh[1][0], Bs[1][0], Wh[1][1], Bs[1][1], Wh[1][2], Bs[1][2], f2);
    k_fps<<<BATCH, 256, 0, stream>>>(x2, 1024, 512, fidx, x3);
    k_ballquery<<<BATCH * 512, 32, 0, stream>>>(x2, 1024, x3, 512, 0.36f, 16, gidx);
    k_sa_mlp<16, 259, 256, 256, 512><<<BATCH * 512, 128, 0, stream>>>(
        x2, 1024, x3, 512, f2, gidx, 1.0f / 0.6f,
        Wh[2][0], Bs[2][0], Wh[2][1], Bs[2][1], Wh[2][2], Bs[2][2], f3);
    k_fps<<<BATCH, 256, 0, stream>>>(x3, 512, 256, fidx, x4);
    k_ballquery<<<BATCH * 256, 32, 0, stream>>>(x3, 512, x4, 256, 1.44f, 8, gidx);
    k_sa_mlp<8, 515, 512, 512, 512><<<BATCH * 256, 128, 0, stream>>>(
        x3, 512, x4, 256, f3, gidx, 1.0f / 1.2f,
        Wh[3][0], Bs[3][0], Wh[3][1], Bs[3][1], Wh[3][2], Bs[3][2], f4);
  };
  encode(noise_pts, nWh, nB, nx1, nf1, nx2, nf2, nx3, nf3, nx4, nf4);
  encode(cond_pts,  cWh, cB, cx1, cf1, cx2, cf2, cx3, cf3, cx4, cf4);

  auto lin = [&](const _Float16* X, const _Float16* W, const float* Bv,
                 _Float16* Y, int M, int K, int O) {
    dim3 g(O / 128, M / 16);
    k_linear_relu<<<g, 128, 0, stream>>>(X, W, Bv, Y, M, K, O);
  };
  auto interp = [&](const float* ux, int nu, const float* kx, int nk,
                    const _Float16* kf, int Ck, const _Float16* uf, int Cu,
                    int addte, _Float16* X) {
    const int total = BATCH * nu;
    k_three_interp<<<(total + 255) / 256, 256, 0, stream>>>(
        ux, nu, kx, nk, kf, Ck, uf, Cu, te, addte, X);
  };

  // ---- FP chain ----
  interp(nx4, 256, cx4, 256, cf4, 512, nf4, 512, 1, xbuf);          // fp1_cross
  lin(xbuf, fWh[0][0], fB[0][0], hbuf, 2048, 1024, 512);
  lin(hbuf, fWh[0][1], fB[0][1], fAb,  2048, 512,  512);
  interp(nx3, 512, nx4, 256, fAb, 512, nf3, 512, 0, xbuf);          // fp1
  lin(xbuf, fWh[1][0], fB[1][0], hbuf, 4096, 1024, 512);
  lin(hbuf, fWh[1][1], fB[1][1], fBb,  4096, 512,  512);
  interp(nx3, 512, cx3, 512, cf3, 512, fBb, 512, 0, xbuf);          // fp2_cross
  lin(xbuf, fWh[2][0], fB[2][0], hbuf, 4096, 1024, 512);
  lin(hbuf, fWh[2][1], fB[2][1], fAb,  4096, 512,  256);
  interp(nx2, 1024, nx3, 512, fAb, 256, nf2, 256, 0, xbuf);         // fp2
  lin(xbuf, fWh[3][0], fB[3][0], hbuf, 8192, 512, 512);
  lin(hbuf, fWh[3][1], fB[3][1], fBb,  8192, 512, 256);
  interp(nx2, 1024, cx2, 1024, cf2, 256, fBb, 256, 0, xbuf);        // fp3_cross
  lin(xbuf, fWh[4][0], fB[4][0], hbuf, 8192, 512, 256);
  lin(hbuf, fWh[4][1], fB[4][1], fAb,  8192, 256, 128);
  interp(nx1, 2048, nx2, 1024, fAb, 128, nf1, 128, 0, xbuf);        // fp3
  lin(xbuf, fWh[5][0], fB[5][0], hbuf, 16384, 256, 256);
  lin(hbuf, fWh[5][1], fB[5][1], fBb,  16384, 256, 128);
  interp(nx1, 2048, cx1, 2048, cf1, 128, fBb, 128, 0, xbuf);        // fp4_cross
  lin(xbuf, fWh[6][0], fB[6][0], hbuf, 16384, 256, 128);
  lin(hbuf, fWh[6][1], fB[6][1], fAb,  16384, 128, 128);
  interp(noise_pts, 8192, nx1, 2048, fAb, 128, nullptr, 0, 0, xbuf); // fp4
  lin(xbuf, fWh[7][0], fB[7][0], hbuf, 65536, 128, 128);
  lin(hbuf, fWh[7][1], fB[7][1], fBb,  65536, 128, 128);

  // ---- LayerNorm + projection ----
  k_ln_out<<<(65536 + 255) / 256, 256, 0, stream>>>(fBb, lnG, lnO, oW,
                                                    (float*)d_out, 65536);
}
